// VectorQuantizer_32727650795529
// MI455X (gfx1250) — compile-verified, matrix operand order checked by reading, not run on hardware
//
#include <hip/hip_runtime.h>
#include <hip/hip_bf16.h>

typedef unsigned short u16;
typedef __attribute__((ext_vector_type(8)))  __bf16 v8bf;
typedef __attribute__((ext_vector_type(16))) __bf16 v16bf;
typedef __attribute__((ext_vector_type(8)))  float  v8f;

#define DDIM 256
#define NCHUNK 32      // codebook rows per LDS chunk
#define CPITCH 264     // padded LDS row pitch in bf16 elems (528 B) -> bank spread

#ifndef __has_builtin
#define __has_builtin(x) 0
#endif

#if __has_builtin(__builtin_amdgcn_global_load_async_to_lds_b128)
#define ASYNC_LDS 1
#else
#define ASYNC_LDS 0
#endif

#if ASYNC_LDS
// Builtin signature (from hipcc diagnostic): (int4 AS1*, int4 AS3*, imm, imm)
typedef __attribute__((ext_vector_type(4))) int int4v;
typedef __attribute__((address_space(1))) int4v* gptr_b128;
typedef __attribute__((address_space(3))) int4v* lptr_b128;

__device__ __forceinline__ gptr_b128 as_gptr(const void* p) {
  // flat global VA == global-AS VA
  return (gptr_b128)(unsigned long long)(uintptr_t)p;
}
__device__ __forceinline__ lptr_b128 as_lptr(void* p) {
  // ISA 10.2: flat LDS address truncates to LDS offset in [31:0]
  return (lptr_b128)(unsigned int)(uintptr_t)p;
}
#endif

template <int N>
__device__ __forceinline__ void wait_async() {
#if ASYNC_LDS
#if __has_builtin(__builtin_amdgcn_s_wait_asynccnt)
  __builtin_amdgcn_s_wait_asynccnt(N);
#else
  asm volatile("s_wait_asynccnt %0" ::"i"(N) : "memory");
#endif
#endif
}

// Build a 16x32-step bf16 WMMA operand fragment for this lane:
// elems 0..7  <- 8 contiguous bf16 at k0 + 8*half        (VGPRs 0-3)
// elems 8..15 <- 8 contiguous bf16 at k0 + 16 + 8*half   (VGPRs 4-7)
// (matches ISA 7.12.2 16-bit A/B wave32 layout)
__device__ __forceinline__ v16bf make_frag(const u16* p) {
  v8bf a = *reinterpret_cast<const v8bf*>(p);
  v8bf b = *reinterpret_cast<const v8bf*>(p + 16);
  return __builtin_shufflevector(a, b, 0,1,2,3,4,5,6,7,8,9,10,11,12,13,14,15);
}

// ---- f32 -> (bf16 hi, bf16 lo) split ------------------------------------
__global__ __launch_bounds__(256) void split_kernel(const float* __restrict__ x,
                                                    u16* __restrict__ hi,
                                                    u16* __restrict__ lo, int n) {
  int i = blockIdx.x * 256 + threadIdx.x;
  if (i < n) {
    float v = x[i];
    __bf16 h = (__bf16)v;
    __bf16 l = (__bf16)(v - (float)h);
    hi[i] = __builtin_bit_cast(u16, h);
    lo[i] = __builtin_bit_cast(u16, l);
  }
}

// ---- exact f32 ||c||^2 per codebook row (1 wave32 per row) ---------------
__global__ __launch_bounds__(256) void cnorm_kernel(const float* __restrict__ cb,
                                                    float* __restrict__ cn, int K) {
  int lane = threadIdx.x & 31;
  int row  = blockIdx.x * 8 + (threadIdx.x >> 5);
  if (row >= K) return;
  float s = 0.f;
  #pragma unroll
  for (int j = 0; j < 8; ++j) {
    float v = cb[(size_t)row * DDIM + lane + 32 * j];
    s += v * v;
  }
  #pragma unroll
  for (int off = 16; off >= 1; off >>= 1) s += __shfl_xor(s, off, 32);
  if (lane == 0) cn[row] = s;
}

// ---- fused bf16x3 WMMA GEMM + argmin -------------------------------------
// Block: 256 threads = 8 waves; each wave owns 16 rows (block = 128 rows).
// A (x) fragments resident in registers; codebook double-buffered through LDS
// with async global->LDS copies (ASYNCcnt) when available.
__global__ __launch_bounds__(256) void vq_argmin_kernel(
    const u16* __restrict__ xh, const u16* __restrict__ xl,
    const u16* __restrict__ ch, const u16* __restrict__ cl,
    const float* __restrict__ cn, int* __restrict__ tok, int K) {
  extern __shared__ u16 smem[];
  const int bufStride = 2 * NCHUNK * CPITCH;   // elems per (H+L) buffer

  const int lane = threadIdx.x & 31;
  const int wave = threadIdx.x >> 5;
  const int half = lane >> 4;
  const int l16  = lane & 15;
  const int rowBase = blockIdx.x * 128;
  const int myRow   = rowBase + wave * 16 + l16;   // A-matrix: M = lane%16

  // Cooperative chunk copy: 32 rows x 256 bf16 (hi & lo), 16B vectors,
  // exactly 8 b128 transfers per thread (4 hi + 4 lo).
  auto issueChunk = [&](int cBase, u16* dH, u16* dL) {
    #pragma unroll
    for (int it = 0; it < 4; ++it) {
      const int v   = threadIdx.x + it * 256;
      const int row = v >> 5;          // /32
      const int c8  = v & 31;
      const u16* gh = ch + (size_t)(cBase + row) * DDIM + c8 * 8;
      const u16* gl = cl + (size_t)(cBase + row) * DDIM + c8 * 8;
      u16* lh = dH + row * CPITCH + c8 * 8;
      u16* ll = dL + row * CPITCH + c8 * 8;
#if ASYNC_LDS
      __builtin_amdgcn_global_load_async_to_lds_b128(as_gptr(gh), as_lptr(lh), 0, 0);
      __builtin_amdgcn_global_load_async_to_lds_b128(as_gptr(gl), as_lptr(ll), 0, 0);
#else
      *(uint4*)lh = *(const uint4*)gh;
      *(uint4*)ll = *(const uint4*)gl;
#endif
    }
  };

  // Resident A fragments: 8 k-steps x (hi,lo) = 128 VGPRs
  v16bf aHi[8], aLo[8];
  #pragma unroll
  for (int kk = 0; kk < 8; ++kk) {
    const size_t base = (size_t)myRow * DDIM + kk * 32 + 8 * half;
    aHi[kk] = make_frag(xh + base);
    aLo[kk] = make_frag(xl + base);
  }

  float best[8];
  int   bestIdx[8];
  #pragma unroll
  for (int r = 0; r < 8; ++r) { best[r] = 3.4e38f; bestIdx[r] = 0; }

  const int nChunks = K / NCHUNK;
  issueChunk(0, smem, smem + NCHUNK * CPITCH);   // prologue: chunk 0 -> buffer 0

  for (int ci = 0; ci < nChunks; ++ci) {
    const int cur = ci & 1;
    u16* curH = smem + cur * bufStride;
    u16* curL = curH + NCHUNK * CPITCH;
    __syncthreads();                    // other buffer fully consumed by all waves
    if (ci + 1 < nChunks) {
      u16* nxtH = smem + (cur ^ 1) * bufStride;
      issueChunk((ci + 1) * NCHUNK, nxtH, nxtH + NCHUNK * CPITCH);
      wait_async<8>();                  // chunk ci's 8 async ops have landed
    } else {
      wait_async<0>();
    }
    __syncthreads();                    // all waves' chunk-ci data visible

    const int cBase = ci * NCHUNK;

    #pragma unroll
    for (int sub = 0; sub < 2; ++sub) {         // two 16-code subtiles
      const int n0 = cBase + sub * 16;
      v8f acc = {};
      #pragma unroll
      for (int kk = 0; kk < 8; ++kk) {          // K = 8 x 32
        const int off = (sub * 16 + l16) * CPITCH + kk * 32 + 8 * half;
        v16bf bHi = make_frag(curH + off);
        v16bf bLo = make_frag(curL + off);
        acc = __builtin_amdgcn_wmma_f32_16x16x32_bf16(false, aHi[kk], false, bHi,
                                                      (short)0, acc, false, false);
        acc = __builtin_amdgcn_wmma_f32_16x16x32_bf16(false, aHi[kk], false, bLo,
                                                      (short)0, acc, false, false);
        acc = __builtin_amdgcn_wmma_f32_16x16x32_bf16(false, aLo[kk], false, bHi,
                                                      (short)0, acc, false, false);
      }
      // epilogue: dist(m,n) ~ ||c||^2 - 2 x.c   (||x||^2 constant per row)
      const float cnv  = cn[n0 + l16];
      const int   idxn = n0 + l16;
      #pragma unroll
      for (int r = 0; r < 8; ++r) {
        float dval = cnv - 2.0f * acc[r];
        bool lt = dval < best[r];
        best[r]    = lt ? dval : best[r];
        bestIdx[r] = lt ? idxn : bestIdx[r];
      }
    }
  }

  // argmin across the 16 N-lanes (xor<16 keeps the two halves independent)
  #pragma unroll
  for (int r = 0; r < 8; ++r) {
    float v = best[r];
    int   i = bestIdx[r];
    #pragma unroll
    for (int s = 8; s >= 1; s >>= 1) {
      float ov = __shfl_xor(v, s, 32);
      int   oi = __shfl_xor(i, s, 32);
      if (ov < v || (ov == v && oi < i)) { v = ov; i = oi; }
    }
    if (l16 == 0) tok[rowBase + wave * 16 + r + 8 * half] = i;  // C-layout row
  }
}

// ---- gather quantized rows + exact MSE accumulation ----------------------
__global__ __launch_bounds__(256) void gather_loss_kernel(
    const float* __restrict__ x, const float* __restrict__ cb,
    const int* __restrict__ tok, float* __restrict__ outq,
    double* lossAcc, int ND) {
  __shared__ float warpSum[8];
  int gid = blockIdx.x * 256 + threadIdx.x;
  float sq = 0.f;
  if (gid < (ND >> 2)) {
    int base = gid << 2;
    int row  = base >> 8;       // /256
    int d    = base & 255;
    int idx  = tok[row];
    float4 q  = *(const float4*)(cb + (size_t)idx * DDIM + d);
    float4 xv = *(const float4*)(x + base);
    *(float4*)(outq + base) = q;
    float a = q.x - xv.x, b = q.y - xv.y, c = q.z - xv.z, e = q.w - xv.w;
    sq = a * a + b * b + c * c + e * e;
  }
  #pragma unroll
  for (int s = 16; s >= 1; s >>= 1) sq += __shfl_xor(sq, s, 32);
  if ((threadIdx.x & 31) == 0) warpSum[threadIdx.x >> 5] = sq;
  __syncthreads();
  if (threadIdx.x == 0) {
    float t = 0.f;
    #pragma unroll
    for (int i = 0; i < 8; ++i) t += warpSum[i];
    atomicAdd(lossAcc, (double)t);
  }
}

// ---- tokens as f32 + usage histogram -------------------------------------
__global__ __launch_bounds__(256) void token_count_kernel(
    const int* __restrict__ tok, float* __restrict__ tokf,
    int* __restrict__ counts, int N) {
  int i = blockIdx.x * 256 + threadIdx.x;
  if (i < N) {
    int t = tok[i];
    tokf[i] = (float)t;
    atomicAdd(&counts[t], 1);
  }
}

// ---- scalar outputs: vq / commitment / codebook losses + perplexity ------
__global__ __launch_bounds__(256) void finalize_kernel(
    const double* __restrict__ lossAcc, const int* __restrict__ counts,
    float* __restrict__ out, int N, int K, int ND) {
  __shared__ double red[256];
  double s = 0.0;
  float invN = 1.0f / (float)N;
  for (int k = threadIdx.x; k < K; k += 256) {
    float avg = (float)counts[k] * invN;
    s += (double)(avg * logf(avg + 1e-10f));
  }
  red[threadIdx.x] = s;
  __syncthreads();
  for (int st = 128; st >= 1; st >>= 1) {
    if (threadIdx.x < st) red[threadIdx.x] += red[threadIdx.x + st];
    __syncthreads();
  }
  if (threadIdx.x == 0) {
    float mse    = (float)(lossAcc[0] / (double)ND);  // e_loss == q_loss
    float commit = 0.25f * mse;
    out[0] = commit + mse;        // vq_loss
    out[1] = commit;              // commitment_loss
    out[2] = mse;                 // codebook_loss
    out[3] = expf(-(float)red[0]);// perplexity
  }
}

extern "C" void kernel_launch(void* const* d_in, const int* in_sizes, int n_in,
                              void* d_out, int out_size, void* d_ws, size_t ws_size,
                              hipStream_t stream) {
  const float* x  = (const float*)d_in[0];   // [32,1024,256] f32
  const float* cb = (const float*)d_in[1];   // [8192,256] f32
  const int ND = in_sizes[0];                // 8388608
  const int N  = ND / DDIM;                  // 32768
  const int KD = in_sizes[1];
  const int K  = KD / DDIM;                  // 8192

  char* ws = (char*)d_ws;
  size_t off = 0;
  auto alloc = [&](size_t bytes) {
    void* p = ws + off;
    off = (off + bytes + 255) & ~(size_t)255;
    return p;
  };
  u16*    xh      = (u16*)alloc((size_t)ND * 2);
  u16*    xl      = (u16*)alloc((size_t)ND * 2);
  u16*    chb     = (u16*)alloc((size_t)KD * 2);
  u16*    clb     = (u16*)alloc((size_t)KD * 2);
  float*  cn      = (float*)alloc((size_t)K * 4);
  int*    tok     = (int*)alloc((size_t)N * 4);
  int*    counts  = (int*)alloc((size_t)K * 4);
  double* lossAcc = (double*)alloc(8);

  float* outq    = (float*)d_out;        // quantized_st [ND]
  float* tokf    = outq + ND;            // tokens as f32 [N]
  float* scalars = tokf + N;             // 4 scalar outputs

  (void)hipMemsetAsync(counts, 0, (size_t)K * 4, stream);
  (void)hipMemsetAsync(lossAcc, 0, 8, stream);

  split_kernel<<<(ND + 255) / 256, 256, 0, stream>>>(x, xh, xl, ND);
  split_kernel<<<(KD + 255) / 256, 256, 0, stream>>>(cb, chb, clb, KD);
  cnorm_kernel<<<(K + 7) / 8, 256, 0, stream>>>(cb, cn, K);

  const size_t smemBytes = (size_t)4 * NCHUNK * CPITCH * sizeof(u16); // 66 KB, dbl-buffered
  vq_argmin_kernel<<<N / 128, 256, smemBytes, stream>>>(xh, xl, chb, clb, cn, tok, K);

  gather_loss_kernel<<<((ND / 4) + 255) / 256, 256, 0, stream>>>(x, cb, tok, outq,
                                                                 lossAcc, ND);
  token_count_kernel<<<(N + 255) / 256, 256, 0, stream>>>(tok, tokf, counts, N);
  finalize_kernel<<<1, 256, 0, stream>>>(lossAcc, counts, scalars, N, K, ND);
}
